// MultiHeadMusicDecoder_11347303596077
// MI455X (gfx1250) — matrix operand order repaired and checked by reading, not compiled, not run on hardware
//
#include <hip/hip_runtime.h>

// MI455X (gfx1250) persistent 2-layer GRU decoder.
// - bf16 WMMA (v_wmma_f32_16x16x32_bf16) for all GEMMs, f32 accumulation
// - 16 workgroups x 256 threads (8 wave32 per WG); 128 waves == 128 16x16 h-tiles
// - weights staged per-block into LDS (48KB), state double-buffered in global ws
// - 3 device-wide barriers per step (atomic arrive + generation spin)

#define H_   256
#define B_   128
#define T_   512
#define NWG  16

typedef __attribute__((ext_vector_type(16))) __bf16 bf16x16;
typedef __attribute__((ext_vector_type(8)))  __bf16 bf16x8;
typedef __attribute__((ext_vector_type(8)))  float  v8f;

__device__ __forceinline__ v8f wmma_bf16(bf16x16 a, bf16x16 b, v8f c) {
  return __builtin_amdgcn_wmma_f32_16x16x32_bf16(
      /*neg_a=*/false, a, /*neg_b=*/false, b,
      /*c_mod=*/(short)0, c, /*reuse_a=*/false, /*reuse_b=*/false);
}

__device__ __forceinline__ float sigmoidf_(float x) {
  return 1.0f / (1.0f + __expf(-x));
}

// A fragment: 16x32 bf16, rows m0..m0+15 of a (B_ x H_) row-major matrix.
// ISA layout: lanes 0-15 hold K = {0..7,16..23}, lanes 16-31 hold K = {8..15,24..31}.
__device__ __forceinline__ bf16x16 load_a(const __bf16* base, int m0, int k0, int lane) {
  const int g = lane >> 4;
  const int m = m0 + (lane & 15);
  const __bf16* p = base + (size_t)m * H_ + k0 + g * 8;
  bf16x8 lo = *(const bf16x8*)(p);        // K = k0 + g*8 + [0..7]
  bf16x8 hi = *(const bf16x8*)(p + 16);   // K = k0 + 16 + g*8 + [0..7]
  bf16x16 r;
#pragma unroll
  for (int i = 0; i < 8; ++i) { r[i] = lo[i]; r[i + 8] = hi[i]; }
  return r;
}

// B fragment: 32x16 bf16 from a 16-row x 256-col block staged in LDS.
// B[k][n] = W[n][k] (weight rows are K-contiguous). Lane n = lane&15 takes
// K = k0 + g*8 + [0..7] and k0 + 16 + g*8 + [0..7], g = lane>=16.
__device__ __forceinline__ bf16x16 load_b(const __bf16* sm, int k0, int lane) {
  const int g = lane >> 4;
  const int n = lane & 15;
  const __bf16* p = sm + n * H_ + k0 + g * 8;
  bf16x8 lo = *(const bf16x8*)(p);
  bf16x8 hi = *(const bf16x8*)(p + 16);
  bf16x16 r;
#pragma unroll
  for (int i = 0; i < 8; ++i) { r[i] = lo[i]; r[i + 8] = hi[i]; }
  return r;
}

// Stage 6 x (16 rows x 256 cols) weight blocks (r/z/n of Wih, r/z/n of Whh).
__device__ __forceinline__ void stage6(__bf16* sm, const __bf16* Wih,
                                       const __bf16* Whh, int j0) {
  const __bf16* srcs[6] = {
    Wih + (size_t)j0 * H_,  Wih + (size_t)(j0 + H_) * H_,  Wih + (size_t)(j0 + 2*H_) * H_,
    Whh + (size_t)j0 * H_,  Whh + (size_t)(j0 + H_) * H_,  Whh + (size_t)(j0 + 2*H_) * H_
  };
#pragma unroll
  for (int it = 0; it < 12; ++it) {
    int i   = (it * 256 + (int)threadIdx.x) * 8;  // element index into 6*4096
    int blk = i >> 12;
    int off = i & 4095;                           // 16 rows x 256 cols contiguous
    *(bf16x8*)(sm + blk * 4096 + off) = *(const bf16x8*)(srcs[blk] + off);
  }
}

__device__ __forceinline__ void stage1(__bf16* sm, const __bf16* Wo, int j0) {
#pragma unroll
  for (int it = 0; it < 2; ++it) {
    int i = (it * 256 + (int)threadIdx.x) * 8;
    *(bf16x8*)(sm + i) = *(const bf16x8*)(Wo + (size_t)j0 * H_ + i);
  }
}

// Device-wide barrier (persistent grid, graph-capture safe).
__device__ __forceinline__ void gridbar(unsigned* bar) {
  __syncthreads();
  if (threadIdx.x == 0) {
    __threadfence();
    unsigned gen  = __hip_atomic_load(&bar[1], __ATOMIC_ACQUIRE, __HIP_MEMORY_SCOPE_AGENT);
    unsigned prev = __hip_atomic_fetch_add(&bar[0], 1u, __ATOMIC_ACQ_REL, __HIP_MEMORY_SCOPE_AGENT);
    if (prev == (unsigned)(NWG - 1)) {
      __hip_atomic_store(&bar[0], 0u, __ATOMIC_RELAXED, __HIP_MEMORY_SCOPE_AGENT);
      __hip_atomic_fetch_add(&bar[1], 1u, __ATOMIC_RELEASE, __HIP_MEMORY_SCOPE_AGENT);
    } else {
      while (__hip_atomic_load(&bar[1], __ATOMIC_ACQUIRE, __HIP_MEMORY_SCOPE_AGENT) == gen) {
        __builtin_amdgcn_s_sleep(2);
      }
    }
    __threadfence();
  }
  __syncthreads();
}

// One fused GRU cell tile: computes h_new[m0:m0+16, j0:j0+16].
// Six WMMA accumulators: (gi,gh) x (r,z,n); K = 256 in 8 steps of 32.
__device__ __forceinline__ void gru_tile(
    const __bf16* xin, const __bf16* hBcur, const float* hFcur,
    float* hFnxt, __bf16* hBnxt, const __bf16* sm,
    const float* brz, const float* bin, const float* bhn,
    int m0, int j0, int lane)
{
  v8f air = {}, aiz = {}, ain_ = {}, ahr = {}, ahz = {}, ahn = {};
#pragma unroll
  for (int k = 0; k < H_; k += 32) {
    bf16x16 ax = load_a(xin,   m0, k, lane);
    bf16x16 ah = load_a(hBcur, m0, k, lane);
    bf16x16 bir = load_b(sm + 0 * 4096, k, lane);
    bf16x16 biz = load_b(sm + 1 * 4096, k, lane);
    bf16x16 bin_ = load_b(sm + 2 * 4096, k, lane);
    bf16x16 bhr = load_b(sm + 3 * 4096, k, lane);
    bf16x16 bhz = load_b(sm + 4 * 4096, k, lane);
    bf16x16 bhn_ = load_b(sm + 5 * 4096, k, lane);
    air  = wmma_bf16(ax, bir,  air);
    aiz  = wmma_bf16(ax, biz,  aiz);
    ain_ = wmma_bf16(ax, bin_, ain_);
    ahr  = wmma_bf16(ah, bhr,  ahr);
    ahz  = wmma_bf16(ah, bhz,  ahz);
    ahn  = wmma_bf16(ah, bhn_, ahn);
  }
  const int j = j0 + (lane & 15);
  const int g = lane >> 4;
  const float b_r  = brz[j];
  const float b_z  = brz[H_ + j];
  const float b_in = bin[j];
  const float b_hn = bhn[j];
#pragma unroll
  for (int r = 0; r < 8; ++r) {
    const int m = m0 + g * 8 + r;           // C/D layout: m = vgpr + 8*(lane>=16)
    float rg = sigmoidf_(air[r] + ahr[r] + b_r);
    float zg = sigmoidf_(aiz[r] + ahz[r] + b_z);
    float ng = tanhf(ain_[r] + b_in + rg * (ahn[r] + b_hn));
    float hv = (1.0f - zg) * ng + zg * hFcur[(size_t)m * H_ + j];
    hFnxt[(size_t)m * H_ + j] = hv;
    hBnxt[(size_t)m * H_ + j] = (__bf16)hv;
  }
}

// Output-head tile: x_next = h1 @ Wo^T + bo; scatter note/dur/gap slices.
__device__ __forceinline__ void head_tile(
    const __bf16* h1B, const __bf16* sm, const float* bo,
    __bf16* xB, float* out, int t, int m0, int j0, int lane)
{
  v8f acc = {};
#pragma unroll
  for (int k = 0; k < H_; k += 32) {
    bf16x16 a = load_a(h1B, m0, k, lane);
    bf16x16 b = load_b(sm, k, lane);
    acc = wmma_bf16(a, b, acc);
  }
  const int j = j0 + (lane & 15);
  const int g = lane >> 4;
  const float bj = bo[j];
  const size_t noteSz = (size_t)B_ * T_ * 128;
  const size_t durSz  = (size_t)B_ * T_ * 64;
#pragma unroll
  for (int r = 0; r < 8; ++r) {
    const int m = m0 + g * 8 + r;           // m = batch index
    float v = acc[r] + bj;
    xB[(size_t)m * H_ + j] = (__bf16)v;
    size_t off;
    if (j < 128)      off = (size_t)m * T_ * 128 + (size_t)t * 128 + j;
    else if (j < 192) off = noteSz + (size_t)m * T_ * 64 + (size_t)t * 64 + (j - 128);
    else              off = noteSz + durSz + (size_t)m * T_ * 64 + (size_t)t * 64 + (j - 192);
    out[off] = v;
  }
}

struct GruP {
  const __bf16 *Wih0, *Whh0, *Wih1, *Whh1, *Wo;
  const float *brz0, *bin0, *bhn0, *brz1, *bin1, *bhn1, *bo;
  __bf16 *xB, *h0B, *h1B;   // h0B/h1B: two B_*H_ buffers back-to-back
  float *h0F, *h1F;
  float *out;
  unsigned *bar;
};

__global__ __launch_bounds__(256, 1) void gru_persist(GruP p) {
  __shared__ __bf16 sm[6 * 16 * H_];        // 48 KB weight stage
  const int lane = threadIdx.x & 31;
  const int wave = threadIdx.x >> 5;
  const int j0 = blockIdx.x * 16;           // block owns one 16-col tile
  const int m0 = wave * 16;                 // wave owns one 16-row tile

  for (int t = 0; t < T_; ++t) {
    const int cur = t & 1, nxt = cur ^ 1;
    // ---- layer 0: h0_new = GRU(x, h0) ----
    stage6(sm, p.Wih0, p.Whh0, j0);
    __syncthreads();
    gru_tile(p.xB, p.h0B + cur * B_ * H_, p.h0F + cur * B_ * H_,
             p.h0F + nxt * B_ * H_, p.h0B + nxt * B_ * H_,
             sm, p.brz0, p.bin0, p.bhn0, m0, j0, lane);
    gridbar(p.bar);
    // ---- layer 1: h1_new = GRU(h0_new, h1) ----
    stage6(sm, p.Wih1, p.Whh1, j0);
    __syncthreads();
    gru_tile(p.h0B + nxt * B_ * H_, p.h1B + cur * B_ * H_, p.h1F + cur * B_ * H_,
             p.h1F + nxt * B_ * H_, p.h1B + nxt * B_ * H_,
             sm, p.brz1, p.bin1, p.bhn1, m0, j0, lane);
    gridbar(p.bar);
    // ---- heads: x_next = h1_new @ Wo^T + bo ----
    stage1(sm, p.Wo, j0);
    __syncthreads();
    head_tile(p.h1B + nxt * B_ * H_, sm, p.bo, p.xB, p.out, t, m0, j0, lane);
    gridbar(p.bar);
  }
}

__global__ void init_bar(unsigned* bar) {
  if (threadIdx.x == 0) { bar[0] = 0u; bar[1] = 0u; }
}

__global__ void prep(
    const float* Wih0, const float* Whh0, const float* bih0, const float* bhh0,
    const float* Wih1, const float* Whh1, const float* bih1, const float* bhh1,
    const float* Wn, const float* bn, const float* Wd, const float* bd,
    const float* Wg, const float* bg, const float* x0,
    __bf16* wih0b, __bf16* whh0b, __bf16* wih1b, __bf16* whh1b, __bf16* wob,
    float* brz0, float* bin0, float* bhn0, float* brz1, float* bin1, float* bhn1,
    float* bo, __bf16* xB, __bf16* h0B, __bf16* h1B, float* h0F, float* h1F)
{
  const int i = blockIdx.x * blockDim.x + threadIdx.x;
  if (i < 3 * H_ * H_) {
    wih0b[i] = (__bf16)Wih0[i]; whh0b[i] = (__bf16)Whh0[i];
    wih1b[i] = (__bf16)Wih1[i]; whh1b[i] = (__bf16)Whh1[i];
  }
  if (i < H_ * H_) {                        // stack Wn/Wd/Wg -> Wo (256x256)
    int j = i >> 8, k = i & 255;
    float v = (j < 128) ? Wn[j * H_ + k]
            : (j < 192) ? Wd[(j - 128) * H_ + k]
                        : Wg[(j - 192) * H_ + k];
    wob[i] = (__bf16)v;
  }
  if (i < 2 * H_) { brz0[i] = bih0[i] + bhh0[i]; brz1[i] = bih1[i] + bhh1[i]; }
  if (i < H_) {
    bin0[i] = bih0[2 * H_ + i]; bhn0[i] = bhh0[2 * H_ + i];
    bin1[i] = bih1[2 * H_ + i]; bhn1[i] = bhh1[2 * H_ + i];
    bo[i] = (i < 128) ? bn[i] : (i < 192) ? bd[i - 128] : bg[i - 192];
  }
  if (i < B_ * H_) {
    xB[i] = (__bf16)x0[i];
    h0B[i] = (__bf16)0.f; h0B[B_ * H_ + i] = (__bf16)0.f;
    h1B[i] = (__bf16)0.f; h1B[B_ * H_ + i] = (__bf16)0.f;
    h0F[i] = 0.f; h0F[B_ * H_ + i] = 0.f;
    h1F[i] = 0.f; h1F[B_ * H_ + i] = 0.f;
  }
}

extern "C" void kernel_launch(void* const* d_in, const int* in_sizes, int n_in,
                              void* d_out, int out_size, void* d_ws, size_t ws_size,
                              hipStream_t stream) {
  (void)in_sizes; (void)n_in; (void)out_size; (void)ws_size;
  const float* Wih0 = (const float*)d_in[3];
  const float* Whh0 = (const float*)d_in[4];
  const float* bih0 = (const float*)d_in[5];
  const float* bhh0 = (const float*)d_in[6];
  const float* Wih1 = (const float*)d_in[7];
  const float* Whh1 = (const float*)d_in[8];
  const float* bih1 = (const float*)d_in[9];
  const float* bhh1 = (const float*)d_in[10];
  const float* Wn   = (const float*)d_in[11];
  const float* bn   = (const float*)d_in[12];
  const float* Wd   = (const float*)d_in[13];
  const float* bd   = (const float*)d_in[14];
  const float* Wg   = (const float*)d_in[15];
  const float* bg   = (const float*)d_in[16];
  const float* x0   = (const float*)d_in[2];   // initial_input (B,1,H)

  size_t off = 0;
  auto alloc = [&](size_t bytes) -> void* {
    off = (off + 255) & ~(size_t)255;
    void* p = (char*)d_ws + off;
    off += bytes;
    return p;
  };
  unsigned* bar   = (unsigned*)alloc(256);
  __bf16* wih0b = (__bf16*)alloc((size_t)3 * H_ * H_ * 2);
  __bf16* whh0b = (__bf16*)alloc((size_t)3 * H_ * H_ * 2);
  __bf16* wih1b = (__bf16*)alloc((size_t)3 * H_ * H_ * 2);
  __bf16* whh1b = (__bf16*)alloc((size_t)3 * H_ * H_ * 2);
  __bf16* wob   = (__bf16*)alloc((size_t)H_ * H_ * 2);
  float* brz0 = (float*)alloc(2 * H_ * 4);
  float* bin0 = (float*)alloc(H_ * 4);
  float* bhn0 = (float*)alloc(H_ * 4);
  float* brz1 = (float*)alloc(2 * H_ * 4);
  float* bin1 = (float*)alloc(H_ * 4);
  float* bhn1 = (float*)alloc(H_ * 4);
  float* bo   = (float*)alloc(H_ * 4);
  __bf16* xB  = (__bf16*)alloc((size_t)B_ * H_ * 2);
  __bf16* h0B = (__bf16*)alloc((size_t)2 * B_ * H_ * 2);
  __bf16* h1B = (__bf16*)alloc((size_t)2 * B_ * H_ * 2);
  float* h0F  = (float*)alloc((size_t)2 * B_ * H_ * 4);
  float* h1F  = (float*)alloc((size_t)2 * B_ * H_ * 4);

  init_bar<<<1, 32, 0, stream>>>(bar);
  prep<<<(3 * H_ * H_ + 255) / 256, 256, 0, stream>>>(
      Wih0, Whh0, bih0, bhh0, Wih1, Whh1, bih1, bhh1,
      Wn, bn, Wd, bd, Wg, bg, x0,
      wih0b, whh0b, wih1b, whh1b, wob,
      brz0, bin0, bhn0, brz1, bin1, bhn1, bo,
      xB, h0B, h1B, h0F, h1F);

  GruP p;
  p.Wih0 = wih0b; p.Whh0 = whh0b; p.Wih1 = wih1b; p.Whh1 = whh1b; p.Wo = wob;
  p.brz0 = brz0; p.bin0 = bin0; p.bhn0 = bhn0;
  p.brz1 = brz1; p.bin1 = bin1; p.bhn1 = bhn1; p.bo = bo;
  p.xB = xB; p.h0B = h0B; p.h1B = h1B; p.h0F = h0F; p.h1F = h1F;
  p.out = (float*)d_out; p.bar = bar;

  gru_persist<<<NWG, 256, 0, stream>>>(p);
}
